// RSSM_82154134438007
// MI455X (gfx1250) — compile-verified, matrix operand order checked by reading, not run on hardware
//
#include <hip/hip_runtime.h>

// ---------------------------------------------------------------------------
// CDNA5 WMMA (wave32): V_WMMA_F32_16X16X32_BF16
// ---------------------------------------------------------------------------
typedef __attribute__((ext_vector_type(16))) __bf16          v16bf;
typedef __attribute__((ext_vector_type(4)))  __bf16          v4bf;
typedef __attribute__((ext_vector_type(8)))  float           v8f;
typedef __attribute__((ext_vector_type(4)))  float           f4;
typedef __attribute__((ext_vector_type(4)))  unsigned short  us4;
typedef __attribute__((ext_vector_type(8)))  unsigned short  us8;

union FragBF { v16bf v; unsigned short u[16]; us8 h[2]; };
union BF4    { v4bf b; us4 u; };

__device__ __forceinline__ unsigned short f2bf(float f) {
  union { __bf16 b; unsigned short u; } x;
  x.b = (__bf16)f;                      // native v_cvt, RNE
  return x.u;
}
__device__ __forceinline__ us4 f2bf4(f4 f) {
  BF4 x;
  x.b = __builtin_convertvector(f, v4bf);
  return x.u;
}

// A tile load+convert+LDS-store with precomputed per-thread offsets.
// base = A + uniform chunk offset; off[i] = loop-invariant vector offset.
template <typename AT, int IT>
__device__ __forceinline__ void loadStoreA(const AT* __restrict__ base,
                                           const int* off, const int* rowv,
                                           const int* colv,
                                           unsigned short (*As)[40]) {
  if constexpr (sizeof(AT) == 4) {
    f4 f[IT];
#pragma unroll
    for (int i = 0; i < IT; ++i) f[i] = *(const f4*)(base + off[i]);
#pragma unroll
    for (int i = 0; i < IT; ++i) *(us4*)&As[rowv[i]][colv[i]] = f2bf4(f[i]);
  } else {
    us4 s[IT];
#pragma unroll
    for (int i = 0; i < IT; ++i) s[i] = *(const us4*)(base + off[i]);
#pragma unroll
    for (int i = 0; i < IT; ++i) *(us4*)&As[rowv[i]][colv[i]] = s[i];
  }
}

// ---------------------------------------------------------------------------
// 2*NSUB WMMAs on one staged K-chunk (wave tile 32 x 16*NSUB)
// ---------------------------------------------------------------------------
template <int NSUB>
__device__ __forceinline__ void computeChunk(
    const unsigned short (*As)[40], const unsigned short (*Bs)[32][16],
    int waveM0, int waveN0, int lane, v8f (&acc)[2][NSUB])
{
  const int half = lane >> 4, lm = lane & 15, kOff = half * 8;
  FragBF a[2], bf[NSUB];
#pragma unroll
  for (int t = 0; t < 2; ++t) {
    int r = waveM0 + t * 16 + lm;
    a[t].h[0] = *(const us8*)&As[r][kOff];        // ds_load_b128
    a[t].h[1] = *(const us8*)&As[r][16 + kOff];
  }
#pragma unroll
  for (int t = 0; t < NSUB; ++t) {
    int st = (waveN0 >> 4) + t;
    bf[t].h[0] = *(const us8*)&Bs[st][lane][0];   // contiguous (swizzled)
    bf[t].h[1] = *(const us8*)&Bs[st][lane][8];
  }
#pragma unroll
  for (int i = 0; i < 2; ++i)
#pragma unroll
    for (int j = 0; j < NSUB; ++j)
      acc[i][j] = __builtin_amdgcn_wmma_f32_16x16x32_bf16(
          false, a[i].v, false, bf[j].v, (short)0, acc[i][j], false, false);
}

// ---------------------------------------------------------------------------
// Generic GEMM: C[M,N] = act( [A1 | A2] @ W + bias ),  W bf16 [K1+K2, N].
// Block tile BMx128, 256 threads (8 wave32s), double-buffered LDS.
// BM=128: 4x2 waves (wave 32x64); BM=64: 2x4 waves (wave 32x32) for 2x the
// workgroup count on the small-batch recurrent GEMMs.
// All per-thread address math hoisted to 32-bit loop-invariant offsets;
// per-chunk addressing is uniform-scalar-base + vector-int-offset.
// ---------------------------------------------------------------------------
template <typename A1T, typename A2T, int MODE, bool HASA2, int BM>
__global__ __launch_bounds__(256) void gemm_bf16_wmma(
    const void* __restrict__ A1v, long a1Stride, int K1,
    const void* __restrict__ A2v, long a2Stride, int K2,
    const unsigned short* __restrict__ W, const float* __restrict__ bias,
    float* __restrict__ Cf, unsigned short* __restrict__ Cb,
    int N, int act)
{
  constexpr int MW   = BM / 32;          // waves along M
  constexpr int NW   = 8 / MW;           // waves along N
  constexpr int NSUB = (128 / NW) / 16;  // 16x16 subtiles per wave along N
  constexpr int IT   = BM / 32;          // A staging groups per thread

  __shared__ unsigned short AsBuf[2][BM][40];      // +8 pad, rows 80B
  __shared__ unsigned short BsBuf[2][8][32][16];   // fragment-order swizzle

  const int tid  = threadIdx.x;
  const int lane = tid & 31;
  const int wave = tid >> 5;
  const int blockN0 = blockIdx.x * 128;
  const int blockM0 = blockIdx.y * BM;
  const int waveM0 = (wave / NW) * 32;
  const int waveN0 = (wave % NW) * (128 / NW);

  const A1T* A1 = (const A1T*)A1v;
  const A2T* A2 = (const A2T*)A2v;

  // ---- loop-invariant per-thread staging offsets (32-bit) ----
  int rowv[IT], colv[IT], a1Off[IT], a2Off[IT];
#pragma unroll
  for (int i = 0; i < IT; ++i) {
    int g = i * 256 + tid;               // BM*8 groups of 4 elements
    rowv[i] = g >> 3;                    // 8 groups per 32-wide row
    colv[i] = (g & 7) * 4;
    int r1 = blockM0 + rowv[i];
    if (MODE == 1) r1 += (r1 >> 6) + 1;  // obs[:,1:] remap, L=64
    a1Off[i] = r1 * (int)a1Stride + colv[i];
    a2Off[i] = (blockM0 + rowv[i]) * (int)a2Stride + colv[i];
  }
  int wOff[2], bSt[2], bLane[2], bIdx[2];
#pragma unroll
  for (int i = 0; i < 2; ++i) {
    int g = i * 256 + tid;               // 512 groups of 8 bf16
    int k = g >> 4, n0 = (g & 15) * 8;
    wOff[i]  = k * N + blockN0 + n0;
    bSt[i]   = n0 >> 4;                  // swizzle: Bs[st][lane][idx]
    bLane[i] = ((k >> 3) & 1) * 16 + (n0 & 15);
    bIdx[i]  = (k & 7) + ((k >> 4) & 1) * 8;
  }

  auto stage = [&](int c, int buf) {
    int kb = c << 5;
    unsigned short (*As)[40]     = AsBuf[buf];
    unsigned short (*Bs)[32][16] = BsBuf[buf];
    us8 w[2];
#pragma unroll
    for (int i = 0; i < 2; ++i)          // W loads first (pipelined)
      w[i] = *(const us8*)(W + (long)kb * N + wOff[i]);
    if (HASA2 && kb >= K1)
      loadStoreA<A2T, IT>(A2 + (kb - K1), a2Off, rowv, colv, As);
    else
      loadStoreA<A1T, IT>(A1 + kb, a1Off, rowv, colv, As);
#pragma unroll
    for (int i = 0; i < 2; ++i)          // scatter into fragment order
#pragma unroll
      for (int e = 0; e < 8; ++e)
        Bs[bSt[i]][bLane[i] + e][bIdx[i]] = w[i][e];
  };

  v8f acc[2][NSUB] = {};
  const int nChunks = (K1 + K2) >> 5;

  stage(0, 0);
  __syncthreads();
  int p = 0;
  for (int c = 0; c < nChunks; ++c) {
    if (c + 1 < nChunks) stage(c + 1, p ^ 1);      // overlap with compute
    if (c + 2 < nChunks)                            // global_prefetch_b8
      __builtin_prefetch((const void*)(W + (long)((c + 2) << 5) * N + blockN0 + (tid >> 1)), 0, 3);
    computeChunk<NSUB>(AsBuf[p], BsBuf[p], waveM0, waveN0, lane, acc);
    __syncthreads();
    p ^= 1;
  }

  // epilogue: C layout VGPR r -> M = r + half*8
  const int half = lane >> 4, lm = lane & 15;
#pragma unroll
  for (int i = 0; i < 2; ++i) {
    int mbase = blockM0 + waveM0 + i * 16 + half * 8;
#pragma unroll
    for (int j = 0; j < NSUB; ++j) {
      int ncol = blockN0 + waveN0 + j * 16 + lm;
      float bv = bias[ncol];
#pragma unroll
      for (int r = 0; r < 8; ++r) {
        float v = acc[i][j][r] + bv;
        if (act == 1) v = (v > 0.f) ? v : (__expf(v) - 1.f);
        long off = (long)(mbase + r) * N + ncol;
        if (Cf) Cf[off] = v; else Cb[off] = f2bf(v);
      }
    }
  }
}

// ---------------------------------------------------------------------------
// GRU gate fusion; also writes deter into both output slots.
// out per (b,l): [pm 256 | ps 256 | pst 256 | det 2048 | qm | qs | qst | det]
// ---------------------------------------------------------------------------
__global__ __launch_bounds__(256) void gru_pointwise(
    const float* __restrict__ gi, const float* __restrict__ gh,
    const float* __restrict__ dprev, float* __restrict__ dnew,
    float* __restrict__ out, int l)
{
  int idx = blockIdx.x * 256 + threadIdx.x;           // 256*2048 threads
  int b = idx >> 11, j = idx & 2047;
  long gb = (long)b * 6144;
  float r = 1.f / (1.f + __expf(-(gi[gb + j]        + gh[gb + j])));
  float z = 1.f / (1.f + __expf(-(gi[gb + 2048 + j] + gh[gb + 2048 + j])));
  float n = tanhf(gi[gb + 4096 + j] + r * gh[gb + 4096 + j]);
  float d = (1.f - z) * n + z * dprev[idx];
  dnew[idx] = d;
  long ob = ((long)b * 64 + l) * 5632;
  out[ob + 768  + j] = d;
  out[ob + 3584 + j] = d;
}

// ---------------------------------------------------------------------------
// Head fusion: std = softplus(raw)+0.1, stoch = mean + std*noise
// ---------------------------------------------------------------------------
__global__ __launch_bounds__(256) void head_pointwise(
    const float* __restrict__ po, const float* __restrict__ noise,
    float* __restrict__ out, int l, int outOff, float* __restrict__ stochOut)
{
  int idx = blockIdx.x * 256 + threadIdx.x;           // 256*256 threads
  int b = idx >> 8, j = idx & 255;
  float mean = po[(long)b * 512 + j];
  float raw  = po[(long)b * 512 + 256 + j];
  float sp   = (raw > 20.f) ? raw : log1pf(__expf(raw));
  float sd   = sp + 0.1f;
  float st   = mean + sd * noise[(long)l * 65536 + (long)b * 256 + j];
  long ob = ((long)b * 64 + l) * 5632 + outOff;
  out[ob + j]       = mean;
  out[ob + 256 + j] = sd;
  out[ob + 512 + j] = st;
  if (stochOut) stochOut[idx] = st;
}

__global__ void cast_f32_to_bf16_v4(const float* __restrict__ src,
                                    unsigned short* __restrict__ dst, long n4) {
  long i = (long)blockIdx.x * blockDim.x + threadIdx.x;
  long stride = (long)gridDim.x * blockDim.x;
  for (; i < n4; i += stride)
    ((us4*)dst)[i] = f2bf4(((const f4*)src)[i]);
}

__global__ void zero_f32(float* __restrict__ p, long n) {
  long i = (long)blockIdx.x * blockDim.x + threadIdx.x;
  long stride = (long)gridDim.x * blockDim.x;
  for (; i < n; i += stride) p[i] = 0.f;
}

// ---------------------------------------------------------------------------
extern "C" void kernel_launch(void* const* d_in, const int* in_sizes, int n_in,
                              void* d_out, int out_size, void* d_ws, size_t ws_size,
                              hipStream_t stream) {
  (void)in_sizes; (void)n_in; (void)out_size; (void)ws_size;
  const float* obs      = (const float*)d_in[0];
  const float* actions  = (const float*)d_in[1];
  const float* noise_pr = (const float*)d_in[2];
  const float* noise_po = (const float*)d_in[3];
  const float* enc_w1 = (const float*)d_in[4];  const float* enc_b1 = (const float*)d_in[5];
  const float* enc_w2 = (const float*)d_in[6];  const float* enc_b2 = (const float*)d_in[7];
  const float* enc_w3 = (const float*)d_in[8];  const float* enc_b3 = (const float*)d_in[9];
  const float* gru_wi = (const float*)d_in[10]; const float* gru_wh = (const float*)d_in[11];
  const float* gru_bi = (const float*)d_in[12]; const float* gru_bh = (const float*)d_in[13];
  const float* prior_w1 = (const float*)d_in[14]; const float* prior_b1 = (const float*)d_in[15];
  const float* prior_w2 = (const float*)d_in[16]; const float* prior_b2 = (const float*)d_in[17];
  const float* post_w1 = (const float*)d_in[18];  const float* post_b1 = (const float*)d_in[19];
  const float* post_w2 = (const float*)d_in[20];  const float* post_b2 = (const float*)d_in[21];
  float* out = (float*)d_out;

  // ---- workspace layout (256B aligned regions) ----
  char* cur = (char*)d_ws;
  auto alloc = [&](size_t bytes) { char* p = cur; cur += (bytes + 255) & ~(size_t)255; return p; };

  unsigned short* wEnc1  = (unsigned short*)alloc(1048576ull  * 2);
  unsigned short* wEnc2  = (unsigned short*)alloc(1048576ull  * 2);
  unsigned short* wEnc3  = (unsigned short*)alloc(1048576ull  * 2);
  unsigned short* wGruWi = (unsigned short*)alloc(1966080ull  * 2);   // 320*6144
  unsigned short* wGruWh = (unsigned short*)alloc(12582912ull * 2);   // 2048*6144
  unsigned short* wPriW1 = (unsigned short*)alloc(2097152ull  * 2);   // 2048*1024
  unsigned short* wPriW2 = (unsigned short*)alloc(524288ull   * 2);   // 1024*512
  unsigned short* wPosW1 = (unsigned short*)alloc(3145728ull  * 2);   // 3072*1024
  unsigned short* wPosW2 = (unsigned short*)alloc(524288ull   * 2);   // 1024*512
  unsigned short* h1     = (unsigned short*)alloc(16777216ull * 2);   // 16384x1024 bf16
  unsigned short* h2     = (unsigned short*)alloc(16777216ull * 2);
  unsigned short* embeds = (unsigned short*)alloc(16777216ull * 2);   // rows b*64+l
  unsigned short* ph     = (unsigned short*)alloc(262144ull   * 2);   // 256x1024 bf16
  unsigned short* qh     = (unsigned short*)alloc(262144ull   * 2);
  float* deterBuf[2] = { (float*)alloc(524288ull * 4), (float*)alloc(524288ull * 4) };
  float* stochBuf[2] = { (float*)alloc(65536ull  * 4), (float*)alloc(65536ull  * 4) };
  float* gi = (float*)alloc(1572864ull * 4);   // 256x6144
  float* gh = (float*)alloc(1572864ull * 4);
  float* pr = (float*)alloc(131072ull  * 4);   // 256x512
  float* qo = (float*)alloc(131072ull  * 4);

  auto cast = [&](const float* s, unsigned short* d, long n) {
    long n4 = n >> 2;
    long blk = (n4 + 255) / 256; if (blk > 2048) blk = 2048;
    cast_f32_to_bf16_v4<<<dim3((unsigned)blk), dim3(256), 0, stream>>>(s, d, n4);
  };

  // ---- weight casts (per-call; L2-resident afterwards: ~48MB bf16 << 192MB) ----
  cast(enc_w1, wEnc1, 1048576);   cast(enc_w2, wEnc2, 1048576);
  cast(enc_w3, wEnc3, 1048576);   cast(gru_wi, wGruWi, 1966080);
  cast(gru_wh, wGruWh, 12582912); cast(prior_w1, wPriW1, 2097152);
  cast(prior_w2, wPriW2, 524288); cast(post_w1, wPosW1, 3145728);
  cast(post_w2, wPosW2, 524288);

  // ---- initial recurrent state = zeros ----
  zero_f32<<<dim3(2048), dim3(256), 0, stream>>>(deterBuf[0], 524288);
  zero_f32<<<dim3(256),  dim3(256), 0, stream>>>(stochBuf[0], 65536);

  // ---- encoder: 3 GEMMs over 16384 rows (BM=128, compute-bound) ----
  gemm_bf16_wmma<float, float, 1, false, 128><<<dim3(8, 128), 256, 0, stream>>>(
      obs, 1024, 1024, nullptr, 0, 0, wEnc1, enc_b1, nullptr, h1, 1024, 1);
  gemm_bf16_wmma<unsigned short, float, 0, false, 128><<<dim3(8, 128), 256, 0, stream>>>(
      h1, 1024, 1024, nullptr, 0, 0, wEnc2, enc_b2, nullptr, h2, 1024, 1);
  gemm_bf16_wmma<unsigned short, float, 0, false, 128><<<dim3(8, 128), 256, 0, stream>>>(
      h2, 1024, 1024, nullptr, 0, 0, wEnc3, enc_b3, nullptr, embeds, 1024, 0);

  // ---- recurrent scan over L = 64 steps (BM=64 -> 2x workgroups) ----
  for (int l = 0; l < 64; ++l) {
    float* dprev = deterBuf[l & 1];
    float* dnew  = deterBuf[(l + 1) & 1];
    float* sprev = stochBuf[l & 1];
    float* snext = stochBuf[(l + 1) & 1];

    // gi = [stoch | action_l] @ gru_wi + gru_bi   (K = 256 + 64)
    gemm_bf16_wmma<float, float, 0, true, 64><<<dim3(48, 4), 256, 0, stream>>>(
        sprev, 256, 256, actions + (long)l * 64, 4096, 64,
        wGruWi, gru_bi, gi, nullptr, 6144, 0);
    // gh = deter @ gru_wh + gru_bh
    gemm_bf16_wmma<float, float, 0, false, 64><<<dim3(48, 4), 256, 0, stream>>>(
        dprev, 2048, 2048, nullptr, 0, 0, wGruWh, gru_bh, gh, nullptr, 6144, 0);
    gru_pointwise<<<dim3(2048), dim3(256), 0, stream>>>(gi, gh, dprev, dnew, out, l);

    // prior head
    gemm_bf16_wmma<float, float, 0, false, 64><<<dim3(8, 4), 256, 0, stream>>>(
        dnew, 2048, 2048, nullptr, 0, 0, wPriW1, prior_b1, nullptr, ph, 1024, 1);
    gemm_bf16_wmma<unsigned short, float, 0, false, 64><<<dim3(4, 4), 256, 0, stream>>>(
        ph, 1024, 1024, nullptr, 0, 0, wPriW2, prior_b2, pr, nullptr, 512, 0);
    head_pointwise<<<dim3(256), dim3(256), 0, stream>>>(pr, noise_pr, out, l, 0, nullptr);

    // posterior head: [deter_new | embed_l] @ post_w1  (K = 2048 + 1024)
    gemm_bf16_wmma<float, unsigned short, 0, true, 64><<<dim3(8, 4), 256, 0, stream>>>(
        dnew, 2048, 2048, embeds + (long)l * 1024, 65536, 1024,
        wPosW1, post_b1, nullptr, qh, 1024, 1);
    gemm_bf16_wmma<unsigned short, float, 0, false, 64><<<dim3(4, 4), 256, 0, stream>>>(
        qh, 1024, 1024, nullptr, 0, 0, wPosW2, post_b2, qo, nullptr, 512, 0);
    head_pointwise<<<dim3(256), dim3(256), 0, stream>>>(qo, noise_po, out, l, 2816, snext);
  }
}
